// LocalAttention_7868380086507
// MI455X (gfx1250) — matrix operand order repaired and checked by reading, not compiled, not run on hardware
//
#include <hip/hip_runtime.h>
#include <hip/hip_bf16.h>

#define HWTOK 12544
#define NTOK  12560
#define CDIM  256
#define NHEAD 8
#define HDIM  32
#define OSEG  7          // key segments for the o_add scan (12544/7 = 1792)

typedef __attribute__((ext_vector_type(16))) __bf16          v16bf;
typedef __attribute__((ext_vector_type(16))) unsigned short  v16us;
typedef __attribute__((ext_vector_type(8)))  float           v8f;

static __device__ inline v8f v8f_zero() {
  v8f z;
#pragma unroll
  for (int i = 0; i < 8; ++i) z[i] = 0.0f;
  return z;
}

// Native fp32 -> bf16 (RNE) via hardware convert
static __device__ inline unsigned short f2bf(float f) {
  __bf16 h = (__bf16)f;
  return __builtin_bit_cast(unsigned short, h);
}
static __device__ inline unsigned pack2bf(float a, float b) {
  return (unsigned)f2bf(a) | ((unsigned)f2bf(b) << 16);
}

static __device__ inline v8f wmma_bf16(v16bf a, v16bf b, v8f c) {
  // D = A(16x32 bf16) x B(32x16 bf16) + C(16x16 f32)
  return __builtin_amdgcn_wmma_f32_16x16x32_bf16(false, a, false, b, (short)0, c,
                                                 false, false);
}

// CDNA5 async global->LDS copy, 16 bytes per lane (ASYNCcnt tracked)
static __device__ inline void async_copy_b128(void* lds_dst, const void* gsrc) {
  unsigned lds = (unsigned)(size_t)lds_dst;  // low 32 bits of flat addr == LDS offset
  asm volatile("global_load_async_to_lds_b128 %0, %1, off"
               :: "v"(lds), "v"(gsrc) : "memory");
}
static __device__ inline void async_wait() {
  asm volatile("s_wait_asynccnt 0" ::: "memory");
}

// A-matrix fragment (16x32 bf16): lane m = lane%16, K split by lane-half,
// VGPR j holds K = (lane>=16?16:0) + 2*j, +1
static __device__ inline v16bf afrag_lds(const unsigned short* src, int ld) {
  const int lane = threadIdx.x & 31;
  const int m = lane & 15;
  const int kb = (lane >> 4) << 4;
  v16us t;
#pragma unroll
  for (int j = 0; j < 8; ++j) {
    const int k = kb + (j << 1);
    t[2 * j]     = src[m * ld + k];
    t[2 * j + 1] = src[m * ld + k + 1];
  }
  return __builtin_bit_cast(v16bf, t);
}

// B-matrix fragment (32x16): B[k][n] = src[n*ld + k]  (src holds N rows of K)
static __device__ inline v16bf bfragT_us(const unsigned short* src, int ld) {
  const int lane = threadIdx.x & 31;
  const int n = lane & 15;
  const int kb = (lane >> 4) << 4;
  v16us t;
#pragma unroll
  for (int j = 0; j < 8; ++j) {
    const int k = kb + (j << 1);
    t[2 * j]     = src[(size_t)n * ld + k];
    t[2 * j + 1] = src[(size_t)n * ld + k + 1];
  }
  return __builtin_bit_cast(v16bf, t);
}

// B fragment from row-major KxN source: B[k][n] = src[k*ld + n]; k>=maxk -> 0
static __device__ inline v16bf bfrag_rm_us(const unsigned short* src, int ld, int maxk) {
  const int lane = threadIdx.x & 31;
  const int n = lane & 15;
  const int kb = (lane >> 4) << 4;
  v16us t;
#pragma unroll
  for (int j = 0; j < 8; ++j) {
    const int k = kb + (j << 1);
    t[2 * j]     = (k     < maxk) ? src[(size_t)k * ld + n]       : (unsigned short)0;
    t[2 * j + 1] = (k + 1 < maxk) ? src[(size_t)(k + 1) * ld + n] : (unsigned short)0;
  }
  return __builtin_bit_cast(v16bf, t);
}

// ---------------------------------------------------------------------------
// K0: one-time fp32 -> bf16 weight conversion (vectorized x4)
// ---------------------------------------------------------------------------
__global__ __launch_bounds__(256) void convert_w_kernel(
    const float* __restrict__ src, unsigned short* __restrict__ dst, int nelem)
{
  const int i = (blockIdx.x * 256 + threadIdx.x) * 4;
  if (i < nelem) {
    const float4 f = *(const float4*)&src[i];
    uint2 p;
    p.x = pack2bf(f.x, f.y);
    p.y = pack2bf(f.z, f.w);
    *(uint2*)&dst[i] = p;
  }
}

// ---------------------------------------------------------------------------
// K1 / K4: QKV GEMM (+ bias, + RoPE for stage 1 spatial q/k), output bf16.
// One block = 16 token rows x ALL 768 columns (8 waves x 6 col-tiles each),
// so x is streamed from HBM exactly once.
// ---------------------------------------------------------------------------
__global__ __launch_bounds__(256) void qkv_rope_kernel(
    const float* __restrict__ X, const unsigned short* __restrict__ Wb,
    const float* __restrict__ Bias, const int* __restrict__ pos2d,
    const unsigned char* __restrict__ mask,
    unsigned short* __restrict__ Qd, unsigned short* __restrict__ Kd,
    unsigned short* __restrict__ Vd,
    unsigned short* __restrict__ Qa, unsigned short* __restrict__ Ka,
    unsigned short* __restrict__ Va,
    int tilesPerBatch, int stage1)
{
  __shared__ __align__(16) unsigned short As[16 * CDIM];
  const int b  = blockIdx.x / tilesPerBatch;
  const int t0 = (blockIdx.x % tilesPerBatch) * 16;
  const int tid = threadIdx.x;
  const size_t rowBase = (size_t)b * NTOK + t0;
  // stage A tile: float4 loads, packed bf16 stores
  for (int i = tid * 4; i < 16 * CDIM; i += 256 * 4) {
    const float4 f = *(const float4*)&X[(rowBase + (i >> 8)) * CDIM + (i & 255)];
    uint2 p;
    p.x = pack2bf(f.x, f.y);
    p.y = pack2bf(f.z, f.w);
    *(uint2*)&As[i] = p;
  }
  __syncthreads();

  const int wave = tid >> 5;
  const int lane = tid & 31;

  v8f acc[6];
#pragma unroll
  for (int ct = 0; ct < 6; ++ct) acc[ct] = v8f_zero();
#pragma unroll
  for (int kk = 0; kk < CDIM / 32; ++kk) {
    v16bf a = afrag_lds(As + kk * 32, CDIM);
#pragma unroll
    for (int ct = 0; ct < 6; ++ct) {
      const int col0 = (wave * 6 + ct) * 16;
      v16bf bm = bfragT_us(Wb + (size_t)col0 * CDIM + kk * 32, CDIM);
      acc[ct] = wmma_bf16(a, bm, acc[ct]);
    }
  }

  const int n    = lane & 15;
  const int half = lane >> 4;
  const float invf = __powf(100.0f, -(float)(n & 7) * 0.125f);

#pragma unroll
  for (int ct = 0; ct < 6; ++ct) {
    const int col0 = (wave * 6 + ct) * 16;
    const int col  = col0 + n;
    const int which = col >> 8;       // 0=q 1=k 2=v
    const int hcol  = col & 255;
    const int head  = hcol >> 5;
    const int d     = hcol & 31;
    const float bias = Bias[col];
    const int pcomp = (d >> 4) & 1;   // first 16 dims use pos.x, next 16 pos.y
    unsigned short* dst  = (which == 0) ? Qd : ((which == 1) ? Kd : Vd);
    unsigned short* dstA = (which == 0) ? Qa : ((which == 1) ? Ka : Va);

#pragma unroll
    for (int r = 0; r < 8; ++r) {
      const int m = r + half * 8;
      const int t = t0 + m;
      float val = acc[ct][r] + bias;
      float pv = __shfl_xor(val, 8, 32);  // rotation partner d <-> d +/- 8
      if (stage1 && which < 2 && t < HWTOK) {
        if (mask[(size_t)b * NTOK + t]) {
          const int pos = pos2d[((size_t)b * NTOK + t) * 2 + pcomp];
          const float ang = (float)pos * invf;
          float s, c;
          __sincosf(ang, &s, &c);
          val = (n < 8) ? (val * c - pv * s) : (val * c + pv * s);
        }
      }
      const unsigned short o = f2bf(val);
      if (stage1) {
        if (t < HWTOK) {
          const int rr = t / 112, cc = t % 112;
          const int win = (b * 16 + rr / 7) * 16 + cc / 7;
          const int tt  = (rr % 7) * 7 + (cc % 7);
          dst[((size_t)(win * NHEAD + head) * 49 + tt) * HDIM + d] = o;
        } else {
          const int a2 = t - HWTOK;
          dstA[(((size_t)b * NHEAD + head) * 16 + a2) * HDIM + d] = o;
        }
      } else {
        dst[(((size_t)b * NHEAD + head) * HWTOK + t) * HDIM + d] = o;
      }
    }
  }
}

// ---------------------------------------------------------------------------
// K2: windowed attention, one WG (128 thr = 4 waves) per (window, head).
// 49 tokens padded to 64; Q/K/V staged via CDNA5 async global->LDS copies.
// ---------------------------------------------------------------------------
__global__ __launch_bounds__(128) void winattn_kernel(
    const unsigned short* __restrict__ Qw, const unsigned short* __restrict__ Kw,
    const unsigned short* __restrict__ Vw, unsigned short* __restrict__ OW)
{
  __shared__ __align__(16) unsigned short Qs[64 * 32];
  __shared__ __align__(16) unsigned short Ks[64 * 32];
  __shared__ __align__(16) unsigned short Vs[64 * 32];
  __shared__ __align__(16) unsigned short Ps[4][16 * 64];
  const int wh = blockIdx.x;
  const int win = wh >> 3, head = wh & 7;
  const size_t base = (size_t)wh * 49 * 32;
  const int tid = threadIdx.x;

  // rows 0..48: async copy from global; rows 49..63: zero-fill (disjoint LDS)
  for (int i = tid * 8; i < 64 * 32; i += 128 * 8) {
    if ((i >> 5) < 49) {
      async_copy_b128(&Qs[i], Qw + base + i);
      async_copy_b128(&Ks[i], Kw + base + i);
      async_copy_b128(&Vs[i], Vw + base + i);
    } else {
      const uint4 z4 = {0u, 0u, 0u, 0u};
      *(uint4*)&Qs[i] = z4;
      *(uint4*)&Ks[i] = z4;
      *(uint4*)&Vs[i] = z4;
    }
  }
  async_wait();
  __syncthreads();

  const int wave = tid >> 5, lane = tid & 31;
  const int m0 = wave * 16;
  const int n = lane & 15, half = lane >> 4;
  const float scale = 0.17677669529663687f;  // 1/sqrt(32)

  v16bf aq = afrag_lds(Qs + m0 * 32, 32);
  v8f s[4];
#pragma unroll
  for (int t2 = 0; t2 < 4; ++t2) {
    v16bf bk = bfragT_us(Ks + t2 * 16 * 32, 32);
    s[t2] = wmma_bf16(aq, bk, v8f_zero());
  }

  float rden[8];
#pragma unroll
  for (int r = 0; r < 8; ++r) {
    float v0[4];
    float mx = -1e30f;
#pragma unroll
    for (int t2 = 0; t2 < 4; ++t2) {
      float x = s[t2][r] * scale;
      if (t2 * 16 + n >= 49) x = -1e30f;  // mask padded keys
      v0[t2] = x;
      mx = fmaxf(mx, x);
    }
#pragma unroll
    for (int off = 1; off < 16; off <<= 1) mx = fmaxf(mx, __shfl_xor(mx, off, 32));
    float sum = 0.0f;
#pragma unroll
    for (int t2 = 0; t2 < 4; ++t2) {
      const float p = __expf(v0[t2] - mx);
      Ps[wave][(r + half * 8) * 64 + t2 * 16 + n] = f2bf(p);
      sum += p;
    }
#pragma unroll
    for (int off = 1; off < 16; off <<= 1) sum += __shfl_xor(sum, off, 32);
    rden[r] = 1.0f / sum;
  }

  const int b = win >> 8, w2 = win & 255;
  const int whh = w2 >> 4, www = w2 & 15;
#pragma unroll
  for (int nt = 0; nt < 2; ++nt) {
    v8f o = v8f_zero();
#pragma unroll
    for (int kk = 0; kk < 2; ++kk) {
      v16bf ap = afrag_lds(&Ps[wave][kk * 32], 64);
      v16bf bv = bfrag_rm_us(Vs + kk * 32 * 32 + nt * 16, 32, 64);
      o = wmma_bf16(ap, bv, o);
    }
#pragma unroll
    for (int r = 0; r < 8; ++r) {
      const int m = m0 + r + half * 8;
      if (m < 49) {
        const int tr = m / 7, tc = m % 7;
        const int tok = (whh * 7 + tr) * 112 + (www * 7 + tc);
        OW[((size_t)b * HWTOK + tok) * CDIM + head * HDIM + nt * 16 + n] =
            f2bf(o[r] * rden[r]);
      }
    }
  }
}

// ---------------------------------------------------------------------------
// K3: output projection GEMM (bf16 x bf16 -> fp32), one block = 16 rows x
// all 256 cols (8 waves x 2 col-tiles); A tile staged with async global->LDS.
// ---------------------------------------------------------------------------
__global__ __launch_bounds__(256) void proj_kernel(
    const unsigned short* __restrict__ Xb, const unsigned short* __restrict__ Wb,
    const float* __restrict__ Bias, float* __restrict__ Out,
    int tilesPerBatch, int inTokStride, int outTokOffset,
    float alpha, int accumulate)
{
  __shared__ __align__(16) unsigned short As[16 * CDIM];
  const int b  = blockIdx.x / tilesPerBatch;
  const int ti = blockIdx.x % tilesPerBatch;
  const size_t inRow  = (size_t)b * inTokStride + (size_t)ti * 16;
  const size_t outRow = (size_t)b * NTOK + outTokOffset + (size_t)ti * 16;
  const int tid = threadIdx.x;
  for (int i = tid * 8; i < 16 * CDIM; i += 256 * 8)
    async_copy_b128(&As[i], Xb + (inRow + (i >> 8)) * CDIM + (i & 255));
  async_wait();
  __syncthreads();

  const int wave = tid >> 5, lane = tid & 31;
  v8f acc[2];
  acc[0] = v8f_zero();
  acc[1] = v8f_zero();
#pragma unroll
  for (int kk = 0; kk < CDIM / 32; ++kk) {
    v16bf a = afrag_lds(As + kk * 32, CDIM);
#pragma unroll
    for (int ct = 0; ct < 2; ++ct) {
      const int col0 = (wave * 2 + ct) * 16;
      v16bf bm = bfragT_us(Wb + (size_t)col0 * CDIM + kk * 32, CDIM);
      acc[ct] = wmma_bf16(a, bm, acc[ct]);
    }
  }
  const int n = lane & 15, half = lane >> 4;
#pragma unroll
  for (int ct = 0; ct < 2; ++ct) {
    const int col = (wave * 2 + ct) * 16 + n;
    const float bias = Bias[col];
#pragma unroll
    for (int r = 0; r < 8; ++r) {
      const int m = r + half * 8;
      const float val = acc[ct][r] + bias;
      float* p = Out + (outRow + m) * CDIM + col;
      if (accumulate) *p += alpha * val; else *p = val;
    }
  }
}

// ---------------------------------------------------------------------------
// K5a: o_add partials — grid (64 bh, OSEG segments); each WG scans 1792 keys
// (8 waves x 7 exact 32-key chunks), online softmax, LDS merge of the 8 wave
// partials, then one un-normalized (M, L, O) partial per (bh, seg) to global.
// ---------------------------------------------------------------------------
__global__ __launch_bounds__(256) void oadd_seg_kernel(
    const unsigned short* __restrict__ Qa, const unsigned short* __restrict__ Ks,
    const unsigned short* __restrict__ Vs,
    float* __restrict__ OmG, float* __restrict__ OlG, float* __restrict__ OoG)
{
  __shared__ __align__(16) unsigned short Qls[16 * 32];
  __shared__ __align__(16) unsigned short Pw[8][16 * 32];
  __shared__ float Oo[8][16 * 32];
  __shared__ float Om[8][16];
  __shared__ float Ol[8][16];
  const int bh = blockIdx.x;
  const int seg = blockIdx.y;
  const int tid = threadIdx.x;
  const int wave = tid >> 5, lane = tid & 31;
  const int n = lane & 15, half = lane >> 4;
  const float scale = 0.17677669529663687f;
  const int SEGKEYS = HWTOK / OSEG;          // 1792
  const int KPW = SEGKEYS / 8;               // 224 = 7 x 32

  if (tid * 8 < 16 * 32) async_copy_b128(&Qls[tid * 8], Qa + (size_t)bh * 512 + tid * 8);
  async_wait();
  __syncthreads();

  v16bf aq = afrag_lds(Qls, 32);
  const size_t kbase = (size_t)bh * HWTOK * HDIM;
  const int k0 = seg * SEGKEYS + wave * KPW;
  float rm[8], rl[8];
  v8f o0 = v8f_zero(), o1 = v8f_zero();
#pragma unroll
  for (int r = 0; r < 8; ++r) { rm[r] = -1e30f; rl[r] = 0.0f; }

  for (int c = 0; c < KPW; c += 32) {
    const size_t krow = (size_t)(k0 + c) * HDIM;
    v16bf b0 = bfragT_us(Ks + kbase + krow, HDIM);
    v16bf b1 = bfragT_us(Ks + kbase + krow + 16 * HDIM, HDIM);
    v8f s0 = wmma_bf16(aq, b0, v8f_zero());
    v8f s1 = wmma_bf16(aq, b1, v8f_zero());
#pragma unroll
    for (int r = 0; r < 8; ++r) {
      const float x0 = s0[r] * scale, x1 = s1[r] * scale;
      float mx = fmaxf(x0, x1);
#pragma unroll
      for (int off = 1; off < 16; off <<= 1) mx = fmaxf(mx, __shfl_xor(mx, off, 32));
      const float mnew = fmaxf(rm[r], mx);
      const float corr = __expf(rm[r] - mnew);
      const float p0 = __expf(x0 - mnew), p1 = __expf(x1 - mnew);
      float ps = p0 + p1;
#pragma unroll
      for (int off = 1; off < 16; off <<= 1) ps += __shfl_xor(ps, off, 32);
      rl[r] = rl[r] * corr + ps;
      rm[r] = mnew;
      o0[r] = o0[r] * corr;
      o1[r] = o1[r] * corr;
      const int row = r + half * 8;
      Pw[wave][row * 32 + n]      = f2bf(p0);
      Pw[wave][row * 32 + 16 + n] = f2bf(p1);
    }
    v16bf ap  = afrag_lds(Pw[wave], 32);
    v16bf bv0 = bfrag_rm_us(Vs + kbase + krow, HDIM, 32);
    v16bf bv1 = bfrag_rm_us(Vs + kbase + krow + 16, HDIM, 32);
    o0 = wmma_bf16(ap, bv0, o0);
    o1 = wmma_bf16(ap, bv1, o1);
  }

#pragma unroll
  for (int r = 0; r < 8; ++r) {
    const int row = r + half * 8;
    Oo[wave][row * 32 + n]      = o0[r];
    Oo[wave][row * 32 + 16 + n] = o1[r];
    if (n == 0) { Om[wave][row] = rm[r]; Ol[wave][row] = rl[r]; }
  }
  __syncthreads();

  if (wave == 0 && lane < 16) {
    const int row = lane;
    float M = -1e30f;
    for (int w = 0; w < 8; ++w) M = fmaxf(M, Om[w][row]);
    float L = 0.0f;
    float accv[32];
    for (int dd = 0; dd < 32; ++dd) accv[dd] = 0.0f;
    for (int w = 0; w < 8; ++w) {
      const float e = __expf(Om[w][row] - M);
      L += Ol[w][row] * e;
      for (int dd = 0; dd < 32; ++dd) accv[dd] += Oo[w][row * 32 + dd] * e;
    }
    const int ps = bh * OSEG + seg;
    OmG[ps * 16 + row] = M;
    OlG[ps * 16 + row] = L;
    for (int dd = 0; dd < 32; ++dd) OoG[(ps * 16 + row) * 32 + dd] = accv[dd];
  }
}

// K5b: merge the OSEG partials per (b,head), normalize, emit bf16 o_add
__global__ __launch_bounds__(32) void oadd_merge_kernel(
    const float* __restrict__ OmG, const float* __restrict__ OlG,
    const float* __restrict__ OoG, unsigned short* __restrict__ OAP)
{
  const int bh = blockIdx.x;
  const int b = bh >> 3, h = bh & 7;
  const int lane = threadIdx.x;
  if (lane < 16) {
    float M = -1e30f;
    for (int s2 = 0; s2 < OSEG; ++s2)
      M = fmaxf(M, OmG[(bh * OSEG + s2) * 16 + lane]);
    float L = 0.0f;
    float accv[32];
    for (int dd = 0; dd < 32; ++dd) accv[dd] = 0.0f;
    for (int s2 = 0; s2 < OSEG; ++s2) {
      const int ps = bh * OSEG + s2;
      const float e = __expf(OmG[ps * 16 + lane] - M);
      L += OlG[ps * 16 + lane] * e;
      for (int dd = 0; dd < 32; ++dd) accv[dd] += OoG[(ps * 16 + lane) * 32 + dd] * e;
    }
    const float inv = 1.0f / L;
    for (int dd = 0; dd < 32; ++dd)
      OAP[((size_t)b * 16 + lane) * CDIM + h * HDIM + dd] = f2bf(accv[dd] * inv);
  }
}

// ---------------------------------------------------------------------------
// K6: upd = attn(Qs[HWTOK], Ka/Va[16]) — one 16-query tile per wave,
// 8 waves = 8 heads per WG.
// ---------------------------------------------------------------------------
__global__ __launch_bounds__(256) void upd_kernel(
    const unsigned short* __restrict__ Qs, const unsigned short* __restrict__ Ka,
    const unsigned short* __restrict__ Va, unsigned short* __restrict__ UPD)
{
  __shared__ __align__(16) unsigned short Pp[8][16 * 32];
  const int b = blockIdx.y;
  const int t0 = blockIdx.x * 16;
  const int tid = threadIdx.x;
  const int head = tid >> 5, lane = tid & 31;
  const int n = lane & 15, half = lane >> 4;
  const float scale = 0.17677669529663687f;
  const size_t qbase = (((size_t)b * NHEAD + head) * HWTOK + t0) * HDIM;
  const size_t abase = ((size_t)b * NHEAD + head) * 16 * HDIM;

  v16bf aq = afrag_lds(Qs + qbase, HDIM);
  v16bf bk = bfragT_us(Ka + abase, HDIM);
  v8f s = wmma_bf16(aq, bk, v8f_zero());

#pragma unroll
  for (int r = 0; r < 8; ++r) {
    const float x = s[r] * scale;
    float mx = x;
#pragma unroll
    for (int off = 1; off < 16; off <<= 1) mx = fmaxf(mx, __shfl_xor(mx, off, 32));
    const float p = __expf(x - mx);
    float sum = p;
#pragma unroll
    for (int off = 1; off < 16; off <<= 1) sum += __shfl_xor(sum, off, 32);
    const int row = r + half * 8;
    Pp[head][row * 32 + n]      = f2bf(p / sum);
    Pp[head][row * 32 + 16 + n] = 0;      // pad K-dim 16..31 with zeros
  }

  v16bf ap  = afrag_lds(Pp[head], 32);
  v16bf bv0 = bfrag_rm_us(Va + abase, HDIM, 16);
  v16bf bv1 = bfrag_rm_us(Va + abase + 16, HDIM, 16);
  v8f o0 = wmma_bf16(ap, bv0, v8f_zero());
  v8f o1 = wmma_bf16(ap, bv1, v8f_zero());

#pragma unroll
  for (int r = 0; r < 8; ++r) {
    const int m = t0 + r + half * 8;
    UPD[((size_t)b * HWTOK + m) * CDIM + head * HDIM + n]      = f2bf(o0[r]);
    UPD[((size_t)b * HWTOK + m) * CDIM + head * HDIM + 16 + n] = f2bf(o1[r]);
  }
}

// ---------------------------------------------------------------------------
extern "C" void kernel_launch(void* const* d_in, const int* in_sizes, int n_in,
                              void* d_out, int out_size, void* d_ws, size_t ws_size,
                              hipStream_t stream)
{
  (void)in_sizes; (void)n_in; (void)out_size; (void)ws_size;
  const float* x            = (const float*)d_in[0];
  const int* pos2d          = (const int*)d_in[1];
  const unsigned char* mask = (const unsigned char*)d_in[2];
  const float* qkv_w        = (const float*)d_in[3];
  const float* qkv_b        = (const float*)d_in[4];
  const float* proj_w       = (const float*)d_in[5];
  const float* proj_b       = (const float*)d_in[6];
  float* out = (float*)d_out;

  char* ws = (char*)d_ws;
  size_t off = 0;
  auto alloc = [&](size_t bytes) -> char* {
    char* p = ws + off;
    off += (bytes + 255) & ~(size_t)255;
    return p;
  };

  const size_t WINE = (size_t)2048 * NHEAD * 49 * HDIM;  // windowed qkv elems
  const size_t SE   = (size_t)8 * NHEAD * HWTOK * HDIM;  // stage-2 qkv elems
  const size_t AE   = (size_t)8 * NHEAD * 16 * HDIM;     // added-token elems
  unsigned short* Qw  = (unsigned short*)alloc(WINE * 2);
  unsigned short* Kw  = (unsigned short*)alloc(WINE * 2);
  unsigned short* Vw  = (unsigned short*)alloc(WINE * 2);
  unsigned short* Qa  = (unsigned short*)alloc(AE * 2);
  unsigned short* Ka  = (unsigned short*)alloc(AE * 2);
  unsigned short* Va  = (unsigned short*)alloc(AE * 2);
  unsigned short* Qs  = (unsigned short*)alloc(SE * 2);
  unsigned short* Ks  = (unsigned short*)alloc(SE * 2);
  unsigned short* Vs  = (unsigned short*)alloc(SE * 2);
  unsigned short* OW  = (unsigned short*)alloc((size_t)8 * HWTOK * CDIM * 2);
  unsigned short* OAP = (unsigned short*)alloc((size_t)8 * 16 * CDIM * 2);
  unsigned short* UPD = (unsigned short*)alloc((size_t)8 * HWTOK * CDIM * 2);
  unsigned short* Wqb = (unsigned short*)alloc((size_t)3 * CDIM * CDIM * 2);
  unsigned short* Wpb = (unsigned short*)alloc((size_t)CDIM * CDIM * 2);
  float* OmG = (float*)alloc((size_t)64 * OSEG * 16 * 4);
  float* OlG = (float*)alloc((size_t)64 * OSEG * 16 * 4);
  float* OoG = (float*)alloc((size_t)64 * OSEG * 16 * 32 * 4);

  // 0) one-time weight conversion to bf16
  convert_w_kernel<<<dim3(3 * CDIM * CDIM / 1024), 256, 0, stream>>>(
      qkv_w, Wqb, 3 * CDIM * CDIM);
  convert_w_kernel<<<dim3(CDIM * CDIM / 1024), 256, 0, stream>>>(
      proj_w, Wpb, CDIM * CDIM);

  // 1) QKV + RoPE for all tokens (stage 1); x streamed once
  qkv_rope_kernel<<<dim3(8 * 785), 256, 0, stream>>>(
      x, Wqb, qkv_b, pos2d, mask, Qw, Kw, Vw, Qa, Ka, Va, 785, 1);
  // 2) windowed attention (2048 windows x 8 heads)
  winattn_kernel<<<dim3(2048 * 8), 128, 0, stream>>>(Qw, Kw, Vw, OW);
  // 3) out-proj -> x2 written into spatial rows of d_out
  proj_kernel<<<dim3(8 * 784), 256, 0, stream>>>(
      OW, Wpb, proj_b, out, 784, HWTOK, 0, 1.0f, 0);
  // 4) QKV of x2 (no RoPE); Qa/Ka/Va untouched (reference recomputes identical values)
  qkv_rope_kernel<<<dim3(8 * 784), 256, 0, stream>>>(
      out, Wqb, qkv_b, pos2d, mask, Qs, Ks, Vs, Qa, Ka, Va, 784, 0);
  // 5) o_add = attn(qa, ks, vs): segmented scan + merge
  oadd_seg_kernel<<<dim3(64, OSEG), 256, 0, stream>>>(Qa, Ks, Vs, OmG, OlG, OoG);
  oadd_merge_kernel<<<dim3(64), 32, 0, stream>>>(OmG, OlG, OoG, OAP);
  // 6) upd = attn(qs, ka, va)
  upd_kernel<<<dim3(784, 8), 256, 0, stream>>>(Qs, Ka, Va, UPD);
  // 7) proj(o_add) -> added rows of d_out
  proj_kernel<<<dim3(8), 256, 0, stream>>>(
      OAP, Wpb, proj_b, out, 1, 16, HWTOK, 1.0f, 0);
  // 8) d_out[:, :HW] += 0.5 * proj(upd)
  proj_kernel<<<dim3(8 * 784), 256, 0, stream>>>(
      UPD, Wpb, proj_b, out, 784, HWTOK, 0, 0.5f, 1);
}
